// Attention_LoRA_20383914787633
// MI455X (gfx1250) — matrix operand order repaired and checked by reading, not compiled, not run on hardware
//
#include <hip/hip_runtime.h>
#include <hip/hip_bf16.h>
#include <math.h>

// ---- problem constants (match reference) ----
#define BB 2
#define SS 2048
#define DD 2048
#define HH 32
#define KVH 8
#define HD 64
#define RR 8
#define NREP (HH / KVH)          // 4
#define MTOT (BB * SS)           // 4096
#define QN (HH * HD)             // 2048
#define KVN (KVH * HD)           // 512
#define LORA_SCALE (0.01f / 8.0f)

typedef __attribute__((ext_vector_type(16))) __bf16         v16bf;
typedef __attribute__((ext_vector_type(8)))  float          v8f;
typedef __attribute__((ext_vector_type(8)))  unsigned short su8;   // 16B of bf16

union BfFrag { v16bf v; su8 h[2]; unsigned short s[16]; };

__device__ __forceinline__ unsigned short f32_to_bf16(float f) {
  union { float f; unsigned u; } x; x.f = f;
  unsigned r = x.u + 0x7FFFu + ((x.u >> 16) & 1u);   // round-to-nearest-even
  return (unsigned short)(r >> 16);
}
__device__ __forceinline__ unsigned pack_bf16(float a, float b) {
  return (unsigned)f32_to_bf16(a) | ((unsigned)f32_to_bf16(b) << 16);
}

// ---------------------------------------------------------------------------
// bulk fp32 -> packed bf16 (n even)
// ---------------------------------------------------------------------------
__global__ void cvt_bf16_kernel(const float* __restrict__ src,
                                unsigned short* __restrict__ dst, long n) {
  long i = (long)blockIdx.x * blockDim.x + threadIdx.x;
  long e = i * 2;
  if (e >= n) return;
  ((unsigned*)dst)[i] = pack_bf16(src[e], src[e + 1]);
}

// ---------------------------------------------------------------------------
// t[m, r] = sum_k X[m,k] * Aw[r,k]   (M x 8, tiny); one block per row
// ---------------------------------------------------------------------------
__global__ void lora_t_kernel(const float* __restrict__ X,
                              const float* __restrict__ Aw,
                              float* __restrict__ T, int K) {
  int m = blockIdx.x;
  float acc[RR];
#pragma unroll
  for (int r = 0; r < RR; ++r) acc[r] = 0.0f;
  for (int k = threadIdx.x; k < K; k += blockDim.x) {
    float xv = X[(size_t)m * K + k];
#pragma unroll
    for (int r = 0; r < RR; ++r) acc[r] += xv * Aw[r * K + k];
  }
#pragma unroll
  for (int off = 16; off > 0; off >>= 1)
#pragma unroll
    for (int r = 0; r < RR; ++r) acc[r] += __shfl_xor(acc[r], off, 32);
  __shared__ float part[8][RR];
  int wid = threadIdx.x >> 5, lane = threadIdx.x & 31;
  if (lane == 0)
#pragma unroll
    for (int r = 0; r < RR; ++r) part[wid][r] = acc[r];
  __syncthreads();
  if (threadIdx.x < RR) {
    float s = 0.0f;
#pragma unroll
    for (int w = 0; w < 8; ++w) s += part[w][threadIdx.x];
    T[(size_t)m * RR + threadIdx.x] = s;
  }
}

// ---------------------------------------------------------------------------
// C[M,N] = Abf[M,K] * Wbf[N,K]^T + scale * (T[M,8] * BL[N,8]^T)
// bf16 inputs preconverted; each wave computes a 16x64 strip (4 n-tiles).
// grid = (N/256, M/16), block = 128 (4 waves)
// ---------------------------------------------------------------------------
__global__ void gemm_bf16_lora(const unsigned short* __restrict__ A,
                               const unsigned short* __restrict__ W,
                               const float* __restrict__ T,
                               const float* __restrict__ BL,
                               float* __restrict__ C,
                               int M, int N, int K, float scale) {
  const int wave = threadIdx.x >> 5;
  const int lane = threadIdx.x & 31;
  const int half = lane >> 4;
  const int l15  = lane & 15;
  const int m0 = blockIdx.y * 16;
  const int nbase = (blockIdx.x * 4 + wave) * 64;
  if (nbase >= N) return;

  const unsigned short* arow = A + (size_t)(m0 + l15) * K + half * 8;
  const unsigned short* wrow[4];
#pragma unroll
  for (int t = 0; t < 4; ++t)
    wrow[t] = W + (size_t)(nbase + t * 16 + l15) * K + half * 8;

  v8f acc[4] = {};
  for (int kb = 0; kb < K; kb += 32) {
    __builtin_prefetch(arow + kb + 256, 0, 1);
    BfFrag af;
    af.h[0] = *(const su8*)(arow + kb);        // K = kb+half*8   .. +7
    af.h[1] = *(const su8*)(arow + kb + 16);   // K = kb+16+half*8.. +7
#pragma unroll
    for (int t = 0; t < 4; ++t) {
      BfFrag bw;
      bw.h[0] = *(const su8*)(wrow[t] + kb);       // K = kb+half*8   .. +7
      bw.h[1] = *(const su8*)(wrow[t] + kb + 16);  // K = kb+16+half*8.. +7
      acc[t] = __builtin_amdgcn_wmma_f32_16x16x32_bf16(
          false, af.v, false, bw.v, (short)0, acc[t], false, false);
    }
  }

  // epilogue: + scale * T[m,:] . BL[n,:]
  float tm[8][RR];
#pragma unroll
  for (int j = 0; j < 8; ++j)
#pragma unroll
    for (int r = 0; r < RR; ++r)
      tm[j][r] = T[(size_t)(m0 + j + 8 * half) * RR + r];
#pragma unroll
  for (int t = 0; t < 4; ++t) {
    const int ncol = nbase + t * 16 + l15;
    float bl[RR];
#pragma unroll
    for (int r = 0; r < RR; ++r) bl[r] = BL[(size_t)ncol * RR + r];
#pragma unroll
    for (int j = 0; j < 8; ++j) {
      float dot = 0.0f;
#pragma unroll
      for (int r = 0; r < RR; ++r) dot += tm[j][r] * bl[r];
      C[(size_t)(m0 + j + 8 * half) * N + ncol] = acc[t][j] + scale * dot;
    }
  }
}

// ---------------------------------------------------------------------------
// RoPE (interleaved pairs) + fp32 -> bf16, with optional scale folded in.
// src: (B*S, nheads*HD) fp32, dst: same shape bf16
// ---------------------------------------------------------------------------
__global__ void rope_cvt_kernel(const float* __restrict__ src,
                                const float* __restrict__ cs,
                                const float* __restrict__ sn,
                                unsigned short* __restrict__ dst,
                                int nheads, float scale) {
  int idx = blockIdx.x * blockDim.x + threadIdx.x;
  int total = MTOT * nheads * (HD / 2);
  if (idx >= total) return;
  int p = idx & 31;
  int h = (idx >> 5) % nheads;
  int m = idx / (32 * nheads);
  int s = m & (SS - 1);
  float c  = cs[s * 32 + p];
  float si = sn[s * 32 + p];
  size_t e = ((size_t)m * nheads + h) * HD + 2 * p;
  float t0 = src[e], t1 = src[e + 1];
  float o0 = (t0 * c - t1 * si) * scale;
  float o1 = (t0 * si + t1 * c) * scale;
  *(unsigned*)(dst + e) = pack_bf16(o0, o1);
}

// ---------------------------------------------------------------------------
// V: (B*S, KVH*HD) fp32  ->  transposed bf16 (B, KVH*HD, S)
// ---------------------------------------------------------------------------
__global__ void vt_cvt_kernel(const float* __restrict__ v,
                              unsigned short* __restrict__ vt) {
  int idx = blockIdx.x * blockDim.x + threadIdx.x;
  int total = BB * KVN * SS;
  if (idx >= total) return;
  int s = idx & (SS - 1);
  int c = (idx >> 11) % KVN;        // SS = 2048 = 2^11
  int b = idx / (SS * KVN);
  vt[idx] = f32_to_bf16(v[((size_t)(b * SS + s)) * KVN + c]);
}

// ---------------------------------------------------------------------------
// Flash attention, causal, GQA, all-bf16 fragments.
// qb: (B*S, H*HD) bf16 (roped, pre-scaled by 0.125)
// kb: (B*S, KVH*HD) bf16 (roped)
// vt: (B, KVH*HD, S) bf16 (transposed)
// ob: (B*S, H*HD) bf16 output
// grid = (S/16, H, B), block = 32 (wave32)
// ---------------------------------------------------------------------------
__global__ void attn_kernel(const unsigned short* __restrict__ qb,
                            const unsigned short* __restrict__ kbuf,
                            const unsigned short* __restrict__ vt,
                            unsigned short* __restrict__ ob) {
  const int lane = threadIdx.x & 31;
  const int half = lane >> 4;
  const int l15  = lane & 15;
  const int m0 = blockIdx.x * 16;
  const int h  = blockIdx.y;
  const int b  = blockIdx.z;
  const int kvh = h / NREP;

  __shared__ __align__(16) unsigned short plds[16][32];   // probs tile, bf16

  // ---- Q fragments: two 16x32 A-frags over head dim ----
  const unsigned short* qrow =
      qb + (size_t)(b * SS + m0 + l15) * QN + h * HD + half * 8;
  BfFrag aq0, aq1;
  aq0.h[0] = *(const su8*)(qrow);
  aq0.h[1] = *(const su8*)(qrow + 16);
  aq1.h[0] = *(const su8*)(qrow + 32);
  aq1.h[1] = *(const su8*)(qrow + 48);

  float mmax[8], lsum[8], corr[8];
  v8f o[4] = {};
#pragma unroll
  for (int j = 0; j < 8; ++j) { mmax[j] = -1e30f; lsum[j] = 0.0f; }

  for (int kv0 = 0; kv0 <= m0 + 15; kv0 += 32) {
    // ---- scores: S(16x32) = Q(16x64) . K^T(64x32) ----
    v8f sf[2];
#pragma unroll
    for (int nt = 0; nt < 2; ++nt) {
      const unsigned short* krow =
          kbuf + (size_t)(b * SS + kv0 + nt * 16 + l15) * KVN
               + kvh * HD + half * 8;
      BfFrag bk0, bk1;
      bk0.h[0] = *(const su8*)(krow);
      bk0.h[1] = *(const su8*)(krow + 16);
      bk1.h[0] = *(const su8*)(krow + 32);
      bk1.h[1] = *(const su8*)(krow + 48);
      v8f z = {};
      z = __builtin_amdgcn_wmma_f32_16x16x32_bf16(false, aq0.v, false, bk0.v,
                                                  (short)0, z, false, false);
      z = __builtin_amdgcn_wmma_f32_16x16x32_bf16(false, aq1.v, false, bk1.v,
                                                  (short)0, z, false, false);
      sf[nt] = z;
    }

    // ---- online softmax per row (row = fixed vgpr j across 16-lane half) ----
#pragma unroll
    for (int j = 0; j < 8; ++j) {
      int m = m0 + j + 8 * half;
      float s0 = sf[0][j] + (((kv0 + l15) > m) ? -1e9f : 0.0f);
      float s1 = sf[1][j] + (((kv0 + 16 + l15) > m) ? -1e9f : 0.0f);
      float mx = fmaxf(s0, s1);
#pragma unroll
      for (int off = 1; off <= 8; off <<= 1) mx = fmaxf(mx, __shfl_xor(mx, off, 32));
      float nm = fmaxf(mmax[j], mx);
      float cr = __expf(mmax[j] - nm);
      float p0 = __expf(s0 - nm);
      float p1 = __expf(s1 - nm);
      float sl = p0 + p1;
#pragma unroll
      for (int off = 1; off <= 8; off <<= 1) sl += __shfl_xor(sl, off, 32);
      lsum[j] = lsum[j] * cr + sl;
      mmax[j] = nm;
      corr[j] = cr;
      plds[j + 8 * half][l15]      = f32_to_bf16(p0);
      plds[j + 8 * half][16 + l15] = f32_to_bf16(p1);
    }
#pragma unroll
    for (int t = 0; t < 4; ++t)
#pragma unroll
      for (int j = 0; j < 8; ++j) o[t][j] *= corr[j];
    __syncthreads();

    // ---- reload P as A-fragment (16x32 bf16) via 16B LDS loads ----
    BfFrag ap;
    ap.h[0] = *(const su8*)&plds[l15][half * 8];
    ap.h[1] = *(const su8*)&plds[l15][16 + half * 8];
    __syncthreads();

    // ---- O(16x64) += P(16x32) . V(32x64), V transposed -> contiguous ----
#pragma unroll
    for (int t = 0; t < 4; ++t) {
      const unsigned short* vrow =
          vt + (size_t)((b * KVH + kvh) * HD + t * 16 + l15) * SS
             + kv0 + half * 8;
      BfFrag bv;
      bv.h[0] = *(const su8*)(vrow);
      bv.h[1] = *(const su8*)(vrow + 16);
      o[t] = __builtin_amdgcn_wmma_f32_16x16x32_bf16(false, ap.v, false, bv.v,
                                                     (short)0, o[t], false, false);
    }
  }

  // ---- normalize + write bf16 out ----
#pragma unroll
  for (int j = 0; j < 8; ++j) {
    float inv = 1.0f / lsum[j];
    size_t row = (size_t)(b * SS + m0 + j + 8 * half) * QN + h * HD;
#pragma unroll
    for (int t = 0; t < 4; ++t)
      ob[row + t * 16 + l15] = f32_to_bf16(o[t][j] * inv);
  }
}

// ---------------------------------------------------------------------------
extern "C" void kernel_launch(void* const* d_in, const int* in_sizes, int n_in,
                              void* d_out, int out_size, void* d_ws, size_t ws_size,
                              hipStream_t stream) {
  const float* x   = (const float*)d_in[0];
  const float* cs  = (const float*)d_in[2];
  const float* sn  = (const float*)d_in[3];
  const float* wq  = (const float*)d_in[5];
  const float* wk  = (const float*)d_in[6];
  const float* wv  = (const float*)d_in[7];
  const float* wo  = (const float*)d_in[8];
  const float* aq  = (const float*)d_in[9];
  const float* bq  = (const float*)d_in[10];
  const float* ak  = (const float*)d_in[11];
  const float* bk  = (const float*)d_in[12];
  const float* av  = (const float*)d_in[13];
  const float* bv  = (const float*)d_in[14];
  const float* ao  = (const float*)d_in[15];
  const float* bo  = (const float*)d_in[16];
  float* y = (float*)d_out;

  // ---- workspace layout (all chunk sizes multiples of 16B) ----
  char* p = (char*)d_ws;
  float* tq  = (float*)p;  p += (size_t)MTOT * RR * 4;
  float* tk  = (float*)p;  p += (size_t)MTOT * RR * 4;
  float* tv  = (float*)p;  p += (size_t)MTOT * RR * 4;
  float* to  = (float*)p;  p += (size_t)MTOT * RR * 4;
  float* xqf = (float*)p;  p += (size_t)MTOT * QN * 4;    // 32MB
  float* xkf = (float*)p;  p += (size_t)MTOT * KVN * 4;   // 8MB
  float* xvf = (float*)p;  p += (size_t)MTOT * KVN * 4;   // 8MB
  unsigned short* xb  = (unsigned short*)p; p += (size_t)MTOT * DD * 2;   // 16MB
  unsigned short* wqb = (unsigned short*)p; p += (size_t)QN * DD * 2;     // 8MB
  unsigned short* wkb = (unsigned short*)p; p += (size_t)KVN * DD * 2;    // 2MB
  unsigned short* wvb = (unsigned short*)p; p += (size_t)KVN * DD * 2;    // 2MB
  unsigned short* wob = (unsigned short*)p; p += (size_t)DD * QN * 2;     // 8MB
  unsigned short* qbb = (unsigned short*)p; p += (size_t)MTOT * QN * 2;   // 16MB
  unsigned short* kbb = (unsigned short*)p; p += (size_t)MTOT * KVN * 2;  // 4MB
  unsigned short* vtb = (unsigned short*)p; p += (size_t)BB * KVN * SS * 2; // 4MB
  unsigned short* obb = (unsigned short*)p; p += (size_t)MTOT * QN * 2;   // 16MB

  // 1) one-time bf16 conversions of activations + weights
  {
    long n;
    n = (long)MTOT * DD;
    cvt_bf16_kernel<<<(n / 2 + 255) / 256, 256, 0, stream>>>(x,  xb,  n);
    n = (long)QN * DD;
    cvt_bf16_kernel<<<(n / 2 + 255) / 256, 256, 0, stream>>>(wq, wqb, n);
    n = (long)KVN * DD;
    cvt_bf16_kernel<<<(n / 2 + 255) / 256, 256, 0, stream>>>(wk, wkb, n);
    cvt_bf16_kernel<<<(n / 2 + 255) / 256, 256, 0, stream>>>(wv, wvb, n);
    n = (long)DD * QN;
    cvt_bf16_kernel<<<(n / 2 + 255) / 256, 256, 0, stream>>>(wo, wob, n);
  }

  // 2) LoRA low-rank projections t = x @ a^T (fp32, tiny)
  lora_t_kernel<<<MTOT, 256, 0, stream>>>(x, aq, tq, DD);
  lora_t_kernel<<<MTOT, 256, 0, stream>>>(x, ak, tk, DD);
  lora_t_kernel<<<MTOT, 256, 0, stream>>>(x, av, tv, DD);
  lora_t_kernel<<<MTOT, 256, 0, stream>>>(x, ao, to, DD);

  // 3) QKV projections (WMMA bf16) with fused LoRA epilogue (fp32 out)
  gemm_bf16_lora<<<dim3(QN / 256, MTOT / 16), 128, 0, stream>>>(
      xb, wqb, tq, bq, xqf, MTOT, QN, DD, LORA_SCALE);
  gemm_bf16_lora<<<dim3(KVN / 256, MTOT / 16), 128, 0, stream>>>(
      xb, wkb, tk, bk, xkf, MTOT, KVN, DD, LORA_SCALE);
  gemm_bf16_lora<<<dim3(KVN / 256, MTOT / 16), 128, 0, stream>>>(
      xb, wvb, tv, bv, xvf, MTOT, KVN, DD, LORA_SCALE);

  // 4) RoPE + bf16 pack (q gets 1/sqrt(HD) folded in); V -> transposed bf16
  {
    int nq = MTOT * HH * (HD / 2);
    int nk = MTOT * KVH * (HD / 2);
    rope_cvt_kernel<<<(nq + 255) / 256, 256, 0, stream>>>(xqf, cs, sn, qbb, HH, 0.125f);
    rope_cvt_kernel<<<(nk + 255) / 256, 256, 0, stream>>>(xkf, cs, sn, kbb, KVH, 1.0f);
    int nv = BB * KVN * SS;
    vt_cvt_kernel<<<(nv + 255) / 256, 256, 0, stream>>>(xvf, vtb);
  }

  // 5) causal GQA flash attention (WMMA bf16), bf16 out
  attn_kernel<<<dim3(SS / 16, HH, BB), 32, 0, stream>>>(qbb, kbb, vtb, obb);

  // 6) output projection + LoRA epilogue -> fp32 d_out
  gemm_bf16_lora<<<dim3(DD / 256, MTOT / 16), 128, 0, stream>>>(
      obb, wob, to, bo, y, MTOT, DD, QN, LORA_SCALE);
}
